// MSA_7318624272436
// MI455X (gfx1250) — compile-verified, matrix operand order checked by reading, not run on hardware
//
#include <hip/hip_runtime.h>
#include <stdint.h>

// MHA for B=2, T=2048, E=1024, H=16, hd=64 on gfx1250 (wave32, WMMA).
// Pass 1: QKV projection -> f16 Q (prescaled by 0.125), K row-major, V
//         TRANSPOSED ([b][h][hd][T]) in d_ws (24 MB).
// Pass 2: flash attention with v_wmma_f32_16x16x32_f16 + fp32 online softmax.
// All WMMA fragments are loaded as two 16B (b128) transfers per fragment.
// K/V tiles are DOUBLE-BUFFERED in LDS and staged with
// GLOBAL_LOAD_ASYNC_TO_LDS_B128 (ASYNCcnt / s_wait_asynccnt), overlapping the
// fetch of block kb+1 with the 8 WMMAs + softmax of block kb.

typedef __attribute__((ext_vector_type(16))) _Float16 v16h;
typedef __attribute__((ext_vector_type(8)))  float    v8f;
typedef __attribute__((ext_vector_type(4)))  float    v4f;
typedef __attribute__((ext_vector_type(4)))  uint32_t v4u;
typedef __attribute__((ext_vector_type(4)))  int      v4i;

#define NUM_HEADS 16
#define EMBED_DIM 1024
#define HEAD_DIM  64
#define SEQ_T     2048
#define BATCH     2
#define N3        192   // 3*HEAD_DIM

#if defined(__HIP_DEVICE_COMPILE__) &&                                   \
    __has_builtin(__builtin_amdgcn_global_load_async_to_lds_b128) &&     \
    __has_builtin(__builtin_amdgcn_s_wait_asynccnt)
#define HAVE_ASYNC 1
#else
#define HAVE_ASYNC 0
#endif

union AF { v16h v; _Float16 h[16]; uint32_t u[8]; v4u q[2]; };

// Load a 16-half WMMA fragment (A or B operand, one 32-wide K step) from a
// row whose K-run is contiguous in memory. Per the CDNA5 ISA layout, lane
// halves 0..7 are K=[lhi*8, lhi*8+8) and halves 8..15 are K=[16+lhi*8, +8):
// two aligned 16-byte loads.
static __device__ __forceinline__ void load_frag16(AF& a,
                                                   const _Float16* row_base,
                                                   int lhi) {
  const v4u* p = (const v4u*)(row_base + lhi * 8);
  a.q[0] = p[0];   // halves 0..7
  a.q[1] = p[2];   // halves 16..23 (p strides 8 halves)
}

// 16-byte global -> LDS copy; async-to-LDS when available.
static __device__ __forceinline__ void cp16(const _Float16* g, _Float16* l) {
#if HAVE_ASYNC
  typedef v4i __attribute__((address_space(1))) gv4i_t;  // global int4*
  typedef v4i __attribute__((address_space(3))) lv4i_t;  // LDS int4*
  __builtin_amdgcn_global_load_async_to_lds_b128(
      (gv4i_t*)(uintptr_t)g, (lv4i_t*)(uint32_t)(uintptr_t)l, 0, 0);
#else
  *(v4u*)l = *(const v4u*)g;
#endif
}

static __device__ __forceinline__ void cp_wait() {
#if HAVE_ASYNC
  __builtin_amdgcn_s_wait_asynccnt(0);
#endif
}

// ---------------------------------------------------------------------------
// Kernel 1: per (b,h) token tile, qkv = x_h @ W + b.  M=tokens, K=64, N=192.
// ---------------------------------------------------------------------------
__global__ __launch_bounds__(256) void qkv_proj_kernel(
    const float* __restrict__ x, const float* __restrict__ W,
    const float* __restrict__ bias, _Float16* __restrict__ Qh,
    _Float16* __restrict__ Kh, _Float16* __restrict__ Vh) {
  __shared__ _Float16 Wt[N3 * HEAD_DIM];  // transposed: [col][k]
  __shared__ float    Bs[N3];

  const int tid  = threadIdx.x;
  const int wave = tid >> 5;
  const int lane = tid & 31;
  const int lhi  = (lane >= 16) ? 1 : 0;
  const int ln   = lane & 15;
  const int tb   = blockIdx.x * 128;
  const int h    = blockIdx.y;
  const int b    = blockIdx.z;

  // Stage W (fp32 [64][192]) transposed as f16 in LDS; coalesced reads.
  for (int idx = tid; idx < HEAD_DIM * N3; idx += 256) {
    int k = idx / N3, c = idx % N3;
    Wt[c * HEAD_DIM + k] = (_Float16)W[idx];
  }
  if (tid < N3) Bs[tid] = bias[tid];
  __syncthreads();

  // A-fragments: x rows (tokens) x hd, loaded as float4 pairs, cvt to f16.
  const int trow = tb + wave * 16 + ln;
  const float* xrow = x + (size_t)(b * SEQ_T + trow) * EMBED_DIM + h * HEAD_DIM;
  AF xa[2];
  for (int ks = 0; ks < 2; ++ks) {
    const v4f* xp = (const v4f*)(xrow + ks * 32 + lhi * 8);
    v4f f0 = xp[0], f1 = xp[1];   // halves 0..7
    v4f f2 = xp[4], f3 = xp[5];   // halves 8..15 (K offset +16)
    for (int e = 0; e < 4; ++e) {
      xa[ks].h[e]      = (_Float16)f0[e];
      xa[ks].h[4 + e]  = (_Float16)f1[e];
      xa[ks].h[8 + e]  = (_Float16)f2[e];
      xa[ks].h[12 + e] = (_Float16)f3[e];
    }
  }

  const size_t hb = (size_t)(b * NUM_HEADS + h) * SEQ_T * HEAD_DIM;

  for (int n = 0; n < 12; ++n) {
    const int colg = n * 16 + ln;
    AF wb0, wb1;
    load_frag16(wb0, Wt + colg * HEAD_DIM, lhi);
    load_frag16(wb1, Wt + colg * HEAD_DIM + 32, lhi);
    float bv = Bs[colg];
    v8f acc;
    for (int e = 0; e < 8; ++e) acc[e] = bv;
    acc = __builtin_amdgcn_wmma_f32_16x16x32_f16(false, xa[0].v, false, wb0.v,
                                                 (short)0, acc, false, false);
    acc = __builtin_amdgcn_wmma_f32_16x16x32_f16(false, xa[1].v, false, wb1.v,
                                                 (short)0, acc, false, false);

    const int cl = (n & 3) * 16 + ln;  // column within the 64-wide head dim
    if (n < 8) {
      _Float16* dst = (n < 4) ? Qh : Kh;
      const float scale = (n < 4) ? 0.125f : 1.0f;  // 1/sqrt(64), exact
      for (int r = 0; r < 8; ++r) {
        int t = tb + wave * 16 + r + (lhi ? 8 : 0);
        dst[hb + (size_t)t * HEAD_DIM + cl] = (_Float16)(acc[r] * scale);
      }
    } else {
      // V stored transposed: [b][h][hd][T]
      for (int r = 0; r < 8; ++r) {
        int t = tb + wave * 16 + r + (lhi ? 8 : 0);
        Vh[hb + (size_t)cl * SEQ_T + t] = (_Float16)acc[r];
      }
    }
  }
}

// ---------------------------------------------------------------------------
// Kernel 2: flash attention. Block = 8 waves = 128 query rows of one (b,h).
// Key blocks of 32: S = Q*K^T (4 WMMA), online softmax, O += P*V (4 WMMA).
// Double-buffered async K/V staging: issue block kb+1, compute block kb.
// ---------------------------------------------------------------------------
__global__ __launch_bounds__(256) void flash_attn_kernel(
    const _Float16* __restrict__ Qh, const _Float16* __restrict__ Kh,
    const _Float16* __restrict__ Vh, float* __restrict__ out) {
  __shared__ _Float16 Ktb[2][32 * HEAD_DIM];  // [key][hd], row-major
  __shared__ _Float16 Vtb[2][HEAD_DIM * 32];  // [hd][key] (global V is [hd][T])
  __shared__ _Float16 Pb[8][16 * 32];         // per-wave P staging

  const int tid  = threadIdx.x;
  const int wave = tid >> 5;
  const int lane = tid & 31;
  const int lhi  = (lane >= 16) ? 1 : 0;
  const int ln   = lane & 15;
  const int qb   = blockIdx.x * 128;
  const int h    = blockIdx.y;
  const int b    = blockIdx.z;
  const size_t hb = (size_t)(b * NUM_HEADS + h) * SEQ_T * HEAD_DIM;

  // Per-thread staging assignments (kb-invariant).
  const int vrow = tid >> 2;          // 0..63 : hd row of V
  const int vseg = tid & 3;           // 0..3  : 8-half segment within 32 keys
  const _Float16* Ksrc0 = Kh + hb + (size_t)tid * 8;        // + kb*2048
  const _Float16* Vsrc0 = Vh + hb + (size_t)vrow * SEQ_T + vseg * 8;  // + kb*32

  // Load this wave's Q A-fragments once (f16, already scaled by 0.125).
  const int trow = qb + wave * 16 + ln;
  const _Float16* qrow = Qh + hb + (size_t)trow * HEAD_DIM;
  AF qa[2];
  load_frag16(qa[0], qrow, lhi);
  load_frag16(qa[1], qrow + 32, lhi);

  float m[8], lsum[8];
  v8f o[4];
  for (int r = 0; r < 8; ++r) { m[r] = -3.0e38f; lsum[r] = 0.0f; }
  for (int d = 0; d < 4; ++d)
    for (int e = 0; e < 8; ++e) o[d][e] = 0.0f;

  _Float16* pw = Pb[wave];

  const int NKB = SEQ_T / 32;  // 64 key blocks

  // Pipeline prologue: issue block 0 into buffer 0.
  cp16(Ksrc0, Ktb[0] + tid * 8);
  cp16(Vsrc0, Vtb[0] + vrow * 32 + vseg * 8);

  for (int kb = 0; kb < NKB; ++kb) {
    const int cur = kb & 1;
    cp_wait();          // this wave's async writes into buf[cur] are done
    __syncthreads();    // everyone's writes done; buf[cur^1] free to refill

    if (kb + 1 < NKB) {  // overlap: fetch next block while computing this one
      cp16(Ksrc0 + (size_t)(kb + 1) * 2048, Ktb[cur ^ 1] + tid * 8);
      cp16(Vsrc0 + (kb + 1) * 32, Vtb[cur ^ 1] + vrow * 32 + vseg * 8);
    }
    if (kb + 2 < NKB) {  // keep L2 warm two blocks ahead (global_prefetch_b8)
      __builtin_prefetch(Ksrc0 + (size_t)(kb + 2) * 2048, 0, 1);
      __builtin_prefetch(Vsrc0 + (kb + 2) * 32, 0, 1);
    }

    const _Float16* Kt = Ktb[cur];
    const _Float16* Vt = Vtb[cur];

    // S = Q * K^T : two 16-key N tiles, each 2 K-steps over hd=64.
    v8f s[2];
    for (int nt = 0; nt < 2; ++nt) {
      const int key = nt * 16 + ln;
      AF f0, f1;
      load_frag16(f0, Kt + key * HEAD_DIM, lhi);
      load_frag16(f1, Kt + key * HEAD_DIM + 32, lhi);
      v8f c;
      for (int e = 0; e < 8; ++e) c[e] = 0.0f;
      c = __builtin_amdgcn_wmma_f32_16x16x32_f16(false, qa[0].v, false, f0.v,
                                                 (short)0, c, false, false);
      c = __builtin_amdgcn_wmma_f32_16x16x32_f16(false, qa[1].v, false, f1.v,
                                                 (short)0, c, false, false);
      s[nt] = c;
    }

    // Online softmax update (fp32). Row stats reduced across 16 lanes.
    float corr[8];
    for (int r = 0; r < 8; ++r) {
      float mx = fmaxf(s[0][r], s[1][r]);
      mx = fmaxf(mx, __shfl_xor(mx, 1));
      mx = fmaxf(mx, __shfl_xor(mx, 2));
      mx = fmaxf(mx, __shfl_xor(mx, 4));
      mx = fmaxf(mx, __shfl_xor(mx, 8));
      float mn = fmaxf(m[r], mx);
      corr[r] = __expf(m[r] - mn);
      m[r] = mn;
      float p0 = __expf(s[0][r] - mn);
      float p1 = __expf(s[1][r] - mn);
      float rs = p0 + p1;
      rs += __shfl_xor(rs, 1);
      rs += __shfl_xor(rs, 2);
      rs += __shfl_xor(rs, 4);
      rs += __shfl_xor(rs, 8);
      lsum[r] = lsum[r] * corr[r] + rs;
      // Stage P (row-major 16x32 f16) for A-fragment reload.
      int row = r + (lhi ? 8 : 0);
      pw[row * 32 + ln]      = (_Float16)p0;
      pw[row * 32 + 16 + ln] = (_Float16)p1;
    }
    for (int d = 0; d < 4; ++d)
      for (int r = 0; r < 8; ++r) o[d][r] *= corr[r];

    // Reload P as a 16x32 A-fragment (DS ops in-order within the wave).
    AF pa;
    load_frag16(pa, pw + ln * 32, lhi);

    // O += P * V : four 16-wide hd tiles, contraction over 32 keys.
    for (int d = 0; d < 4; ++d) {
      AF vf;
      load_frag16(vf, Vt + (d * 16 + ln) * 32, lhi);
      o[d] = __builtin_amdgcn_wmma_f32_16x16x32_f16(false, pa.v, false, vf.v,
                                                    (short)0, o[d], false, false);
    }
  }

  // Normalize and write fp32 output [B, T, E].
  for (int r = 0; r < 8; ++r) {
    float inv = 1.0f / lsum[r];
    int t = qb + wave * 16 + r + (lhi ? 8 : 0);
    float* orow = out + (size_t)(b * SEQ_T + t) * EMBED_DIM + h * HEAD_DIM;
    for (int d = 0; d < 4; ++d) orow[d * 16 + ln] = o[d][r] * inv;
  }
}

// ---------------------------------------------------------------------------
extern "C" void kernel_launch(void* const* d_in, const int* in_sizes, int n_in,
                              void* d_out, int out_size, void* d_ws, size_t ws_size,
                              hipStream_t stream) {
  (void)in_sizes; (void)n_in; (void)out_size; (void)ws_size;
  const float* x    = (const float*)d_in[0];
  const float* W    = (const float*)d_in[1];
  const float* bias = (const float*)d_in[2];

  const size_t elems = (size_t)BATCH * NUM_HEADS * SEQ_T * HEAD_DIM;  // 4 Mi
  _Float16* Qh = (_Float16*)d_ws;          // 8 MB, [b][h][t][hd]
  _Float16* Kh = Qh + elems;               // 8 MB, [b][h][t][hd]
  _Float16* Vh = Kh + elems;               // 8 MB, [b][h][hd][t] (transposed)

  dim3 grid(SEQ_T / 128, NUM_HEADS, BATCH);  // 16 x 16 x 2 = 512 blocks
  qkv_proj_kernel<<<grid, 256, 0, stream>>>(x, W, bias, Qh, Kh, Vh);
  flash_attn_kernel<<<grid, 256, 0, stream>>>(Qh, Kh, Vh, (float*)d_out);
}
